// GINSY_62783831933364
// MI455X (gfx1250) — matrix operand order repaired and checked by reading, not compiled, not run on hardware
//
#include <hip/hip_runtime.h>
#include <hip/hip_bf16.h>
#include <stdint.h>

#define N_NODES 50000
#define N_EDGES 800000
#define IN_DIM  512
#define HID     256

typedef __attribute__((ext_vector_type(16))) __bf16 v16bf;
typedef __attribute__((ext_vector_type(8)))  float  v8f;
typedef __attribute__((ext_vector_type(4)))  int    v4i;

union FragAB { uint4 u[2]; v16bf v; };

__device__ __forceinline__ unsigned short f32_to_bf16_rne(float f) {
    union { float f; uint32_t u; } v; v.f = f;
    uint32_t r = v.u + 0x7FFFu + ((v.u >> 16) & 1u);
    return (unsigned short)(r >> 16);
}

// ---- async global->LDS staging (gfx1250 path, with safe fallback) ----
#if __has_builtin(__builtin_amdgcn_global_load_async_to_lds_b128)
#define HAVE_ASYNC_LDS 1
#else
#define HAVE_ASYNC_LDS 0
#endif

__device__ __forceinline__ void stage_b128(const unsigned short* g, unsigned short* l) {
#if HAVE_ASYNC_LDS
    __builtin_amdgcn_global_load_async_to_lds_b128(
        (__attribute__((address_space(1))) v4i*)g,
        (__attribute__((address_space(3))) v4i*)l, 0, 0);
#else
    *(uint4*)l = *(const uint4*)g;   // global_load_b128 + ds_store_b128
#endif
}

__device__ __forceinline__ void stage_wait() {
#if HAVE_ASYNC_LDS
    asm volatile("s_wait_asynccnt 0" ::: "memory");
#endif
    __syncthreads();
}

// ---------------- elementwise helpers ----------------

__global__ void copy_f32v4(const float4* __restrict__ in, float4* __restrict__ out, long n4) {
    for (long i = (long)blockIdx.x * blockDim.x + threadIdx.x; i < n4;
         i += (long)gridDim.x * blockDim.x)
        out[i] = in[i];
}

__global__ void cast_f32_to_bf16(const float* __restrict__ in, unsigned short* __restrict__ out, long n) {
    for (long i = (long)blockIdx.x * blockDim.x + threadIdx.x; i < n;
         i += (long)gridDim.x * blockDim.x)
        out[i] = f32_to_bf16_rne(in[i]);
}

// W is K x N row-major f32; Wt is N x K row-major bf16 (i.e. transposed)
__global__ void transpose_to_bf16(const float* __restrict__ W, unsigned short* __restrict__ Wt,
                                  int K, int N) {
    int total = K * N;
    for (int i = blockIdx.x * blockDim.x + threadIdx.x; i < total;
         i += gridDim.x * blockDim.x) {
        int k = i / N;
        int n = i - k * N;
        Wt[(size_t)n * K + k] = f32_to_bf16_rne(W[i]);
    }
}

// agg[dst] += x[src] ; agg pre-initialized with x (fuses the "+ x_i" of GIN eps=0)
__global__ void scatter_add_edges(const float* __restrict__ x,
                                  const int* __restrict__ src,
                                  const int* __restrict__ dst,
                                  float* __restrict__ agg, int D) {
    int D4 = D >> 2;
    long total = (long)N_EDGES * D4;
    for (long i = (long)blockIdx.x * blockDim.x + threadIdx.x; i < total;
         i += (long)gridDim.x * blockDim.x) {
        int e = (int)(i / D4);
        int c = (int)(i - (long)e * D4);
        int s = src[e], d = dst[e];
        float4 v = ((const float4*)(x + (size_t)s * D))[c];
        float* base = agg + (size_t)d * D + (size_t)c * 4;
        atomicAdd(base + 0, v.x);
        atomicAdd(base + 1, v.y);
        atomicAdd(base + 2, v.z);
        atomicAdd(base + 3, v.w);
    }
}

// per-row L2 normalize (D == HID == 256, one block per row)
__global__ void l2norm_rows(const float* __restrict__ in, float* __restrict__ out) {
    __shared__ float sred[HID];
    int row = blockIdx.x;
    int t = threadIdx.x;
    float v = in[(size_t)row * HID + t];
    sred[t] = v * v;
    __syncthreads();
    for (int s = HID / 2; s > 0; s >>= 1) {
        if (t < s) sred[t] += sred[t + s];
        __syncthreads();
    }
    float norm = fmaxf(sqrtf(sred[0]), 1e-12f);
    out[(size_t)row * HID + t] = v / norm;
}

// ---------------- WMMA bf16 GEMM with LDS-staged B ----------------
// C(M x Nc) = A(M x K, bf16 row-major) * B(K x Nc), B supplied transposed as
// Bt (Nc x K bf16). Block = 8 waves = 128(M) x 64(N) tile; the block async-
// copies its 64 x K B-strip into LDS once (rows padded by 8 elements to
// spread banks), then each wave computes a 16x64 strip: per 32-K step it
// loads one A fragment from global (2x b128) + 4 B fragments from LDS
// (8x ds_load_b128) and issues 4 back-to-back WMMAs.
// EPI: 0 = bias+relu -> bf16, 1 = bias+relu -> f32,
//      2 = bias+elu  -> bf16, 3 = bias only  -> f32.
template <int EPI>
__global__ void gemm_wmma_bf16(const unsigned short* __restrict__ A,
                               const unsigned short* __restrict__ Bt,
                               const float* __restrict__ bias,
                               void* __restrict__ out,
                               int K, int Nc, int mTiles) {
    extern __shared__ unsigned short ldsB[];      // 64 rows x (K+8) bf16
    const int KP = K + 8;
    int groups_n = Nc >> 6;                       // 64-wide N strips
    int bm = blockIdx.x / groups_n;
    int bn = blockIdx.x - bm * groups_n;

    // ---- stage B strip (64 x K) into LDS ----
    const unsigned short* __restrict__ Bg = Bt + (size_t)(bn * 64) * K;
    int chunksPerRow = K >> 3;                    // 16B chunks per row
    int totalChunks = chunksPerRow << 6;          // 64 rows
    for (int c = threadIdx.x; c < totalChunks; c += blockDim.x) {
        int r  = c / chunksPerRow;
        int ci = c - r * chunksPerRow;
        stage_b128(Bg + (size_t)r * K + (size_t)ci * 8,
                   ldsB + (size_t)r * KP + (size_t)ci * 8);
    }
    stage_wait();

    int waveId = threadIdx.x >> 5;
    int lane = threadIdx.x & 31;
    int mt = bm * 8 + waveId;
    if (mt >= mTiles) return;                     // wave-uniform tail guard
    int half = lane >> 4;
    int li   = lane & 15;

    const unsigned short* __restrict__ Arow = A + (size_t)(mt * 16 + li) * K;
    const unsigned short* lBrow = ldsB + (size_t)li * KP;

    v8f acc[4];
#pragma unroll
    for (int j = 0; j < 4; ++j) acc[j] = (v8f){0.f, 0.f, 0.f, 0.f, 0.f, 0.f, 0.f, 0.f};

    for (int k0 = 0; k0 < K; k0 += 32) {
        FragAB a;
        // A 16x32 bf16 fragment: elems 0-7  -> K = k0 + half*8 + (0..7)
        //                        elems 8-15 -> K = k0 + 16 + half*8 + (0..7)
        a.u[0] = *(const uint4*)(Arow + k0 + half * 8);
        a.u[1] = *(const uint4*)(Arow + k0 + 16 + half * 8);
        __builtin_prefetch(Arow + k0 + 64, 0, 1);
        FragAB b[4];
#pragma unroll
        for (int j = 0; j < 4; ++j) {
            // row j*16+li of the LDS strip, K = k0 + half*8 + (0..7), k0 + 16 + half*8 + (0..7)
            const unsigned short* bp = lBrow + (size_t)j * 16 * KP + k0;
            b[j].u[0] = *(const uint4*)(bp + half * 8);
            b[j].u[1] = *(const uint4*)(bp + 16 + half * 8);
        }
#pragma unroll
        for (int j = 0; j < 4; ++j)
            acc[j] = __builtin_amdgcn_wmma_f32_16x16x32_bf16(
                /*neg_a=*/false, a.v, /*neg_b=*/false, b[j].v,
                /*c_mod=*/(short)0, acc[j], /*reuse_a=*/false, /*reuse_b=*/false);
    }

#pragma unroll
    for (int j = 0; j < 4; ++j) {
        int col = bn * 64 + j * 16 + li;
        float bv = bias[col];
#pragma unroll
        for (int v = 0; v < 8; ++v) {
            int rowo = mt * 16 + half * 8 + v;   // C layout: VGPR v holds M = v + half*8
            float x = acc[j][v] + bv;
            if (EPI == 0 || EPI == 1) x = fmaxf(x, 0.f);
            if (EPI == 2) x = (x > 0.f) ? x : (__expf(x) - 1.f);
            if (EPI == 0 || EPI == 2)
                ((unsigned short*)out)[(size_t)rowo * Nc + col] = f32_to_bf16_rne(x);
            else
                ((float*)out)[(size_t)rowo * Nc + col] = x;
        }
    }
}

// ---------------- host-side orchestration ----------------

static inline char* carve(char*& p, size_t bytes) {
    char* r = p;
    p += (bytes + 255) & ~(size_t)255;
    return r;
}

static inline void launch_gemm_all(hipStream_t stream, int epi,
                                   const unsigned short* A, const unsigned short* Bt,
                                   const float* bias, void* out, int K, int Nc) {
    const int EW = 256;
    int mTiles = N_NODES / 16;                       // 3125
    int gblocks = ((mTiles + 7) / 8) * (Nc >> 6);    // 391 * strips
    size_t shmem = (size_t)64 * (K + 8) * sizeof(unsigned short);
    switch (epi) {
    case 0: gemm_wmma_bf16<0><<<gblocks, EW, shmem, stream>>>(A, Bt, bias, out, K, Nc, mTiles); break;
    case 1: gemm_wmma_bf16<1><<<gblocks, EW, shmem, stream>>>(A, Bt, bias, out, K, Nc, mTiles); break;
    case 2: gemm_wmma_bf16<2><<<gblocks, EW, shmem, stream>>>(A, Bt, bias, out, K, Nc, mTiles); break;
    default: gemm_wmma_bf16<3><<<gblocks, EW, shmem, stream>>>(A, Bt, bias, out, K, Nc, mTiles); break;
    }
}

static void run_gin_layer(hipStream_t stream,
                          const float* xin, int K,
                          const int* src, const int* dst,
                          const float* w1, const float* b1,
                          const float* w2, const float* b2,
                          float* agg, unsigned short* Abf, unsigned short* Hbf,
                          float* Hf, float* hout,
                          unsigned short* wtA, unsigned short* wtB) {
    const int EW = 256;
    long n  = (long)N_NODES * K;
    long n4 = n / 4;

    // agg = x ; agg[dst] += x[src]
    copy_f32v4<<<2048, EW, 0, stream>>>((const float4*)xin, (float4*)agg, n4);
    scatter_add_edges<<<4096, EW, 0, stream>>>(xin, src, dst, agg, K);
    // A = bf16(x + agg)
    cast_f32_to_bf16<<<2048, EW, 0, stream>>>(agg, Abf, n);
    // weights -> transposed bf16
    transpose_to_bf16<<<512, EW, 0, stream>>>(w1, wtA, K, HID);
    transpose_to_bf16<<<512, EW, 0, stream>>>(w2, wtB, HID, HID);
    // H1 = relu(A @ w1 + b1)  (bf16)
    launch_gemm_all(stream, 0, Abf, wtA, b1, Hbf, K, HID);
    // H2 = relu(H1 @ w2 + b2) (f32)
    launch_gemm_all(stream, 1, Hbf, wtB, b2, Hf, HID, HID);
    // h = l2_normalize(H2)
    l2norm_rows<<<N_NODES, HID, 0, stream>>>(Hf, hout);
}

extern "C" void kernel_launch(void* const* d_in, const int* in_sizes, int n_in,
                              void* d_out, int out_size, void* d_ws, size_t ws_size,
                              hipStream_t stream) {
    (void)in_sizes; (void)n_in; (void)out_size; (void)ws_size;

    const float* x     = (const float*)d_in[0];
    const int*   edges = (const int*)d_in[1];
    const int*   src   = edges;
    const int*   dst   = edges + N_EDGES;
    const float* g0_w1 = (const float*)d_in[2],  *g0_b1 = (const float*)d_in[3];
    const float* g0_w2 = (const float*)d_in[4],  *g0_b2 = (const float*)d_in[5];
    const float* g1_w1 = (const float*)d_in[6],  *g1_b1 = (const float*)d_in[7];
    const float* g1_w2 = (const float*)d_in[8],  *g1_b2 = (const float*)d_in[9];
    const float* g2_w1 = (const float*)d_in[10], *g2_b1 = (const float*)d_in[11];
    const float* g2_w2 = (const float*)d_in[12], *g2_b2 = (const float*)d_in[13];
    const float* l1_w  = (const float*)d_in[14], *l1_b  = (const float*)d_in[15];
    const float* l2_w  = (const float*)d_in[16], *l2_b  = (const float*)d_in[17];

    char* p = (char*)d_ws;
    float*          agg = (float*)         carve(p, (size_t)N_NODES * IN_DIM * sizeof(float));
    unsigned short* Abf = (unsigned short*)carve(p, (size_t)N_NODES * IN_DIM * sizeof(unsigned short));
    unsigned short* Hbf = (unsigned short*)carve(p, (size_t)N_NODES * HID * sizeof(unsigned short));
    float*          Hf  = (float*)         carve(p, (size_t)N_NODES * HID * sizeof(float));
    float*          h   = (float*)         carve(p, (size_t)N_NODES * HID * sizeof(float));
    unsigned short* wtA = (unsigned short*)carve(p, (size_t)IN_DIM * HID * sizeof(unsigned short));
    unsigned short* wtB = (unsigned short*)carve(p, (size_t)HID * HID * sizeof(unsigned short));

    // three GIN layers (h lives in f32 between layers)
    run_gin_layer(stream, x, IN_DIM, src, dst, g0_w1, g0_b1, g0_w2, g0_b2,
                  agg, Abf, Hbf, Hf, h, wtA, wtB);
    run_gin_layer(stream, h, HID,    src, dst, g1_w1, g1_b1, g1_w2, g1_b2,
                  agg, Abf, Hbf, Hf, h, wtA, wtB);
    run_gin_layer(stream, h, HID,    src, dst, g2_w1, g2_b1, g2_w2, g2_b2,
                  agg, Abf, Hbf, Hf, h, wtA, wtB);

    // final MLP head: elu(h @ l1_w + l1_b) @ l2_w + l2_b
    const int EW = 256;
    long nh = (long)N_NODES * HID;
    cast_f32_to_bf16<<<2048, EW, 0, stream>>>(h, Abf, nh);
    transpose_to_bf16<<<512, EW, 0, stream>>>(l1_w, wtA, HID, HID);
    launch_gemm_all(stream, 2, Abf, wtA, l1_b, Hbf, HID, HID);
    transpose_to_bf16<<<512, EW, 0, stream>>>(l2_w, wtB, HID, HID);
    launch_gemm_all(stream, 3, Hbf, wtB, l2_b, d_out, HID, HID);
}
